// DynamicsSolver_51522427683093
// MI455X (gfx1250) — matrix operand order repaired and checked by reading, not compiled, not run on hardware
//
#include <hip/hip_runtime.h>
#include <hip/hip_bf16.h>
#include <math.h>

// ---------------------------------------------------------------------------
// CDNA5 (gfx1250) wave32 WMMA implementation of the GNN dynamics solver.
// All 128-wide GEMMs run on v_wmma_f32_16x16x32_f16 (f16 A/B, f32 accum).
// Weights are pre-packed in B-fragment order; activations are stored in
// A-fragment-shuffled order so every WMMA operand is a contiguous v16h load.
// ---------------------------------------------------------------------------

typedef __attribute__((ext_vector_type(16))) _Float16 v16h;
typedef __attribute__((ext_vector_type(8)))  float    v8f;

#define EPSF 1e-7f

// ---------------- small float3 helpers ----------------
struct F3 { float x, y, z; };
__device__ __forceinline__ F3 operator+(F3 a, F3 b) { return {a.x+b.x, a.y+b.y, a.z+b.z}; }
__device__ __forceinline__ F3 operator-(F3 a, F3 b) { return {a.x-b.x, a.y-b.y, a.z-b.z}; }
__device__ __forceinline__ F3 operator*(F3 a, float s) { return {a.x*s, a.y*s, a.z*s}; }
__device__ __forceinline__ float dot3(F3 a, F3 b) { return a.x*b.x + a.y*b.y + a.z*b.z; }
__device__ __forceinline__ F3 cross3(F3 a, F3 b) {
  return {a.y*b.z - a.z*b.y, a.z*b.x - a.x*b.z, a.x*b.y - a.y*b.x};
}
__device__ __forceinline__ F3 nrm3(F3 v) {
  float n = sqrtf(dot3(v, v));
  float s = 1.f / fmaxf(n, EPSF);
  return v * s;
}
__device__ __forceinline__ F3 ld3(const float* p, int i) {
  return {p[i*3+0], p[i*3+1], p[i*3+2]};
}

// ---------------------------------------------------------------------------
// A-fragment shuffle: within-row offset so that lane l's 16 halves for a
// 32-wide K block are contiguous.  ISA A layout (16-bit, 16x32):
//   lane<16 : element e -> K = ((e>>3)<<4) + (e&7)
//   lane>=16: element e -> K = ((e>>3)<<4) + 8 + (e&7)
// ashuf places K index c at: kblock*32 + hibit*16 + e
__device__ __forceinline__ int ashuf(int c) {
  return (c & ~31) + ((c >> 3) & 1) * 16 + ((c >> 4) & 1) * 8 + (c & 7);
}

// ---------------------------------------------------------------------------
// WMMA tile GEMM: Y[16x128] = act(X[16xK] @ W[Kx128] + b)
//  X: A-shuffled rows (row stride xstride halves), one v16h load per fragment
//  Wp: packed B fragments [(kb*8+t)*32 + lane][16]
//  YSHUF=true : Y written f16 in A-shuffled order (feeds next GEMM)
//  YSHUF=false: Y written row-major (LN staging / dot-product consumers)
template<bool YSHUF, typename TY>
__device__ __forceinline__ void wgemm_s(const _Float16* Xs, int xstride, int K,
                                        const _Float16* Wp, const float* bias,
                                        TY* Y, bool relu, int lane) {
  const int n   = lane & 15;
  const int m   = lane & 15;
  const int hb  = (lane >> 4) & 1;
  const int hiA = hb ? 8 : 0;
  v8f acc[8];
#pragma unroll
  for (int t = 0; t < 8; ++t) {
    float bv = bias[t*16 + n];
#pragma unroll
    for (int v = 0; v < 8; ++v) acc[t][v] = bv;
  }
  const int KB = K >> 5;
  for (int kb = 0; kb < KB; ++kb) {
    v16h a = *(const v16h*)(Xs + m * xstride + kb*32 + hb*16);
#pragma unroll
    for (int t = 0; t < 8; ++t) {
      v16h b = *(const v16h*)(Wp + (size_t)((kb*8 + t)*32 + lane) * 16);
      acc[t] = __builtin_amdgcn_wmma_f32_16x16x32_f16(
          false, a, false, b, (short)0, acc[t], false, false);
    }
  }
#pragma unroll
  for (int t = 0; t < 8; ++t)
#pragma unroll
    for (int v = 0; v < 8; ++v) {
      float y = acc[t][v];
      if (relu) y = fmaxf(y, 0.f);
      const int M = v + hiA;
      if (YSHUF) Y[M*128 + ashuf(t*16 + n)] = (TY)y;
      else       Y[M*128 + t*16 + n]        = (TY)y;
    }
}

// LayerNorm stats with 2 lanes per row (lane and lane+16 each sum 64 cols).
__device__ __forceinline__ void ln_stats2(const float* rowp, int half,
                                          float& mean, float& inv) {
  float s = 0.f;
  for (int j = 0; j < 64; ++j) s += rowp[half*64 + j];
  s += __shfl_xor(s, 16, 32);
  float m = s * (1.f / 128.f);
  float v = 0.f;
  for (int j = 0; j < 64; ++j) { float d = rowp[half*64 + j] - m; v += d * d; }
  v += __shfl_xor(v, 16, 32);
  mean = m;
  inv = rsqrtf(v * (1.f / 128.f) + 1e-5f);
}

// ---------------- weight pack (f32 row-major -> f16 B-fragment order) -------
__global__ void k_pack(const float* __restrict__ src, _Float16* __restrict__ dst,
                       int K, int Kpad) {
  int i = blockIdx.x * blockDim.x + threadIdx.x;
  if (i >= Kpad * 128) return;
  int e    = i & 15;
  int lane = (i >> 4) & 31;
  int t    = (i >> 9) & 7;
  int kb   = i >> 12;
  int k = kb*32 + ((lane & 16) ? 8 : 0) + ((e >> 3) << 4) + (e & 7);
  int n = t*16 + (lane & 15);
  float v = (k < K) ? src[k*128 + n] : 0.f;
  dst[i] = (_Float16)v;
}

__global__ void k_zero(float* p, int n) {
  int i = blockIdx.x * blockDim.x + threadIdx.x;
  if (i < n) p[i] = 0.f;
}

// ---------------- node encoder: node_feat[N,6] -> node_latent (LN, f16) -----
__global__ void __launch_bounds__(32)
k_node_latent(const float* __restrict__ node_feat, int N,
              const _Float16* w1, const float* b1,
              const _Float16* w2, const float* b2,
              const float* g, const float* beta,
              _Float16* __restrict__ nl16a) {
  __shared__ alignas(32) _Float16 x[16*32];
  __shared__ alignas(32) _Float16 h[16*128];
  __shared__ alignas(32) float    o[16*128];
  const int lane = threadIdx.x;
  const int row0 = blockIdx.x * 16;
  for (int i = lane; i < 16*32; i += 32) {
    int r = i >> 5, c = i & 31;
    float v = 0.f;
    if (c < 6 && row0 + r < N) v = node_feat[(size_t)(row0 + r) * 6 + c];
    x[r*32 + ashuf(c)] = (_Float16)v;
  }
  __syncthreads();
  wgemm_s<true, _Float16>(x, 32, 32, w1, b1, h, true, lane);
  __syncthreads();
  wgemm_s<false, float>(h, 128, 128, w2, b2, o, false, lane);
  __syncthreads();
  {
    const int row = lane & 15, half = lane >> 4;
    float m, inv; ln_stats2(&o[row*128], half, m, inv);
    if (row0 + row < N) {
      size_t base = (size_t)(row0 + row) * 128;
      for (int j = 0; j < 64; ++j) {
        int c = half*64 + j;
        float y = (o[row*128 + c] - m) * inv * g[c] + beta[c];
        nl16a[base + ashuf(c)] = (_Float16)y;
      }
    }
  }
}

// ---------------- per-node heads: nw (sigmoid), mi, ii, de ------------------
__global__ void __launch_bounds__(32)
k_node_head(const _Float16* __restrict__ nl16a, int N,
            const _Float16* nw_w1, const float* nw_b1, const float* nw_w2, const float* nw_b2,
            const _Float16* mi_w1, const float* mi_b1, const float* mi_w2, const float* mi_b2,
            const _Float16* ii_w1, const float* ii_b1, const float* ii_w2, const float* ii_b2,
            const _Float16* de_w1, const float* de_b1, const float* de_w2, const float* de_b2,
            float* __restrict__ w_sig, float* __restrict__ m_inv,
            float* __restrict__ i_inv, float* __restrict__ de_o) {
  __shared__ alignas(32) _Float16 h[16*128];
  const int lane = threadIdx.x;
  const int row0 = blockIdx.x * 16;
  const _Float16* X = nl16a + (size_t)row0 * 128;
  const int row = lane & 15, half = lane >> 4;
  const bool wr = (half == 0) && (row0 + row < N);

  wgemm_s<false, _Float16>(X, 128, 128, nw_w1, nw_b1, h, true, lane);
  __syncthreads();
  {
    float z = 0.f;
    for (int k = 0; k < 64; ++k) z += (float)h[row*128 + half*64 + k] * nw_w2[half*64 + k];
    z += __shfl_xor(z, 16, 32);
    if (wr) w_sig[row0 + row] = 1.f / (1.f + expf(-(z + nw_b2[0])));
  }
  __syncthreads();

  wgemm_s<false, _Float16>(X, 128, 128, mi_w1, mi_b1, h, true, lane);
  __syncthreads();
  {
    float z = 0.f;
    for (int k = 0; k < 64; ++k) z += (float)h[row*128 + half*64 + k] * mi_w2[half*64 + k];
    z += __shfl_xor(z, 16, 32);
    if (wr) m_inv[row0 + row] = z + mi_b2[0];
  }
  __syncthreads();

  wgemm_s<false, _Float16>(X, 128, 128, ii_w1, ii_b1, h, true, lane);
  __syncthreads();
  {
    float z = 0.f;
    for (int k = 0; k < 64; ++k) z += (float)h[row*128 + half*64 + k] * ii_w2[half*64 + k];
    z += __shfl_xor(z, 16, 32);
    if (wr) i_inv[row0 + row] = z + ii_b2[0];
  }
  __syncthreads();

  wgemm_s<false, _Float16>(X, 128, 128, de_w1, de_b1, h, true, lane);
  __syncthreads();
#pragma unroll
  for (int j = 0; j < 3; ++j) {
    float z = 0.f;
    for (int k = 0; k < 64; ++k) z += (float)h[row*128 + half*64 + k] * de_w2[(half*64 + k)*3 + j];
    z += __shfl_xor(z, 16, 32);
    if (wr) de_o[(size_t)(row0 + row) * 3 + j] = z + de_b2[j];
  }
}

// ---------------- fused edge encoder: geometry + ef(s)+ef(r)+ee + ie --------
__global__ void __launch_bounds__(32)
k_edge_enc(const int* __restrict__ ei, int E, int N,
           const float* __restrict__ pos, const float* __restrict__ vel,
           const float* __restrict__ pvel, const float* __restrict__ omg,
           const float* __restrict__ pomg, const float* __restrict__ acl,
           const float* __restrict__ alp, const float* __restrict__ eattr,
           const _Float16* ef_w1, const float* ef_b1, const _Float16* ef_w2, const float* ef_b2,
           const float* ef_g, const float* ef_b,
           const _Float16* ee_w1, const float* ee_b1, const _Float16* ee_w2, const float* ee_b2,
           const float* ee_g, const float* ee_b,
           const _Float16* ie_w1, const float* ie_b1, const _Float16* ie_w2, const float* ie_b2,
           const float* ie_g, const float* ie_b,
           const _Float16* __restrict__ nl16a,
           float* __restrict__ basis, float* __restrict__ il_out,
           _Float16* __restrict__ il16a) {
  __shared__ alignas(32) _Float16 sf[16*32];
  __shared__ alignas(32) _Float16 rf[16*32];
  __shared__ alignas(32) _Float16 efb[16*32];
  __shared__ alignas(32) _Float16 h[16*128];
  __shared__ alignas(32) float    o[16*128];
  __shared__ alignas(32) _Float16 sr16[16*128];
  __shared__ alignas(32) _Float16 el16[16*128];
  __shared__ int sidx[16], ridx[16];

  const int lane = threadIdx.x;
  const int row0 = blockIdx.x * 16;
  const int row  = lane & 15, half = lane >> 4;

  // ---- per-edge reference frame + projected features (lanes 0..15) ----
  if (lane < 16) {
    const int eF = row0 + row;
    const int e = (eF < E) ? eF : (E - 1);
    const int s = ei[e], r = ei[E + e];
    sidx[row] = s; ridx[row] = r;
    F3 sp = ld3(pos, s),  rp = ld3(pos, r);
    F3 sv = ld3(vel, s),  rv = ld3(vel, r);
    F3 spv = ld3(pvel, s), rpv = ld3(pvel, r);
    F3 so = ld3(omg, s),  ro = ld3(omg, r);
    F3 spo = ld3(pomg, s), rpo = ld3(pomg, r);
    F3 sa = ld3(acl, s),  ra = ld3(acl, r);
    F3 sal = ld3(alp, s), ral = ld3(alp, r);

    F3 rel = rp - sp;
    float dn = sqrtf(dot3(rel, rel));
    float dist = fmaxf(dn, EPSF);
    F3 va = rel * (1.f / dist);
    F3 b1v = nrm3(cross3(rv - sv, va));
    F3 b2v = nrm3(sv + rv);
    F3 b3v = nrm3(cross3(ro - so, va));
    F3 b4v = nrm3(so + ro);
    F3 bs = b1v + b2v + b3v + b4v;
    F3 bprl = va * dot3(bs, va);
    F3 vb = nrm3(cross3(bs - bprl, va));
    F3 vc = nrm3(cross3(bprl, vb));

    if (eF < E) {
      float* bb = basis + (size_t)eF * 9;
      bb[0]=va.x; bb[1]=va.y; bb[2]=va.z;
      bb[3]=vb.x; bb[4]=vb.y; bb[5]=vb.z;
      bb[6]=vc.x; bb[7]=vc.y; bb[8]=vc.z;
    }

    auto P = [&](F3 v, int base, _Float16* buf, float sgn) {
      buf[row*32 + ashuf(base + 0)] = (_Float16)(sgn * dot3(va, v));
      buf[row*32 + ashuf(base + 1)] = (_Float16)(sgn * dot3(vb, v));
      buf[row*32 + ashuf(base + 2)] = (_Float16)(sgn * dot3(vc, v));
    };
    P(sv, 0, sf, 1.f); P(spv, 3, sf, 1.f); P(so, 6, sf, 1.f);
    P(spo, 9, sf, 1.f); P(sa, 12, sf, 1.f); P(sal, 15, sf, 1.f);
    P(rv, 0, rf, -1.f); P(rpv, 3, rf, -1.f); P(ro, 6, rf, -1.f);
    P(rpo, 9, rf, 1.f); P(ra, 12, rf, -1.f); P(ral, 15, rf, 1.f);
    for (int j = 18; j < 32; ++j) {
      sf[row*32 + ashuf(j)] = (_Float16)0.f;
      rf[row*32 + ashuf(j)] = (_Float16)0.f;
    }
    efb[row*32 + ashuf(0)] = (_Float16)dn;
    efb[row*32 + ashuf(1)] = (_Float16)eattr[e*3 + 0];
    efb[row*32 + ashuf(2)] = (_Float16)eattr[e*3 + 1];
    efb[row*32 + ashuf(3)] = (_Float16)eattr[e*3 + 2];
    for (int j = 4; j < 32; ++j) efb[row*32 + ashuf(j)] = (_Float16)0.f;
  }
  __syncthreads();

  // ---- s_lat = LN(ef(sf)) -> sr16 (A-shuffled) ----
  wgemm_s<true, _Float16>(sf, 32, 32, ef_w1, ef_b1, h, true, lane);
  __syncthreads();
  wgemm_s<false, float>(h, 128, 128, ef_w2, ef_b2, o, false, lane);
  __syncthreads();
  {
    float m, inv; ln_stats2(&o[row*128], half, m, inv);
    for (int j = 0; j < 64; ++j) {
      int c = half*64 + j;
      sr16[row*128 + ashuf(c)] =
          (_Float16)((o[row*128 + c] - m) * inv * ef_g[c] + ef_b[c]);
    }
  }
  __syncthreads();

  // ---- r_lat = LN(ef(rf)); sr16 += r_lat ----
  wgemm_s<true, _Float16>(rf, 32, 32, ef_w1, ef_b1, h, true, lane);
  __syncthreads();
  wgemm_s<false, float>(h, 128, 128, ef_w2, ef_b2, o, false, lane);
  __syncthreads();
  {
    float m, inv; ln_stats2(&o[row*128], half, m, inv);
    for (int j = 0; j < 64; ++j) {
      int c = half*64 + j;
      float y = (o[row*128 + c] - m) * inv * ef_g[c] + ef_b[c];
      sr16[row*128 + ashuf(c)] = (_Float16)((float)sr16[row*128 + ashuf(c)] + y);
    }
  }
  __syncthreads();

  // ---- edge_latent = LN(ee(efb)) -> el16 (A-shuffled) ----
  wgemm_s<true, _Float16>(efb, 32, 32, ee_w1, ee_b1, h, true, lane);
  __syncthreads();
  wgemm_s<false, float>(h, 128, 128, ee_w2, ee_b2, o, false, lane);
  __syncthreads();
  {
    float m, inv; ln_stats2(&o[row*128], half, m, inv);
    for (int j = 0; j < 64; ++j) {
      int c = half*64 + j;
      el16[row*128 + ashuf(c)] =
          (_Float16)((o[row*128 + c] - m) * inv * ee_g[c] + ee_b[c]);
    }
  }
  __syncthreads();

  // ---- ie layer1: K=384, A = [sr16 | nl16a[s]+nl16a[r] | el16] ----
  {
    const int n  = lane & 15;
    const int m  = lane & 15;
    const int hb = (lane >> 4) & 1;
    const int hiA = hb ? 8 : 0;
    v8f accv[8];
#pragma unroll
    for (int t = 0; t < 8; ++t) {
      float bv = ie_b1[t*16 + n];
#pragma unroll
      for (int v = 0; v < 8; ++v) accv[t][v] = bv;
    }
#pragma unroll
    for (int kb = 0; kb < 12; ++kb) {
      v16h a;
      if (kb < 4) {
        a = *(const v16h*)(sr16 + m*128 + kb*32 + hb*16);
      } else if (kb < 8) {
        const int off = (kb - 4)*32 + hb*16;
        v16h as = *(const v16h*)(nl16a + (size_t)sidx[m]*128 + off);
        v16h ar = *(const v16h*)(nl16a + (size_t)ridx[m]*128 + off);
        a = as + ar;  // packed f16 add
      } else {
        a = *(const v16h*)(el16 + m*128 + (kb - 8)*32 + hb*16);
      }
#pragma unroll
      for (int t = 0; t < 8; ++t) {
        v16h b = *(const v16h*)(ie_w1 + (size_t)((kb*8 + t)*32 + lane) * 16);
        accv[t] = __builtin_amdgcn_wmma_f32_16x16x32_f16(
            false, a, false, b, (short)0, accv[t], false, false);
      }
    }
#pragma unroll
    for (int t = 0; t < 8; ++t)
#pragma unroll
      for (int v = 0; v < 8; ++v)
        h[(v + hiA)*128 + ashuf(t*16 + n)] = (_Float16)fmaxf(accv[t][v], 0.f);
  }
  __syncthreads();

  // ---- ie layer2 + LN -> il (f32 out) + il16a (A-shuffled f16) ----
  wgemm_s<false, float>(h, 128, 128, ie_w2, ie_b2, o, false, lane);
  __syncthreads();
  {
    float m, inv; ln_stats2(&o[row*128], half, m, inv);
    if (row0 + row < E) {
      size_t base = (size_t)(row0 + row) * 128;
      for (int j = 0; j < 64; ++j) {
        int c = half*64 + j;
        float y = (o[row*128 + c] - m) * inv * ie_g[c] + ie_b[c];
        il_out[base + c] = y;
        il16a[base + ashuf(c)] = (_Float16)y;
      }
    }
  }
}

// ---------------- edge decoder: il -> fij/tauij, scatter-add ---------------
__global__ void __launch_bounds__(32)
k_edge_dec(const int* __restrict__ ei, int E,
           const _Float16* __restrict__ il16a, const float* __restrict__ pos,
           const float* __restrict__ basis, const float* __restrict__ w_sig,
           const _Float16* i1_w1, const float* i1_b1, const float* i1_w2, const float* i1_b2,
           const _Float16* i2_w1, const float* i2_b1, const float* i2_w2, const float* i2_b2,
           const _Float16* fs_w1, const float* fs_b1, const float* fs_w2, const float* fs_b2,
           float* __restrict__ out_f, float* __restrict__ out_t) {
  __shared__ alignas(32) _Float16 h[16*128];
  const int lane = threadIdx.x;
  const int row0 = blockIdx.x * 16;
  const int row = lane & 15, half = lane >> 4;
  const _Float16* X = il16a + (size_t)row0 * 128;
  float cf0, cf1, cf2, ca0, ca1, ca2, lam;

  wgemm_s<false, _Float16>(X, 128, 128, i1_w1, i1_b1, h, true, lane);
  __syncthreads();
  {
    float z0 = 0.f, z1 = 0.f, z2 = 0.f;
    for (int k = 0; k < 64; ++k) {
      float hv = (float)h[row*128 + half*64 + k];
      const float* w = &i1_w2[(half*64 + k)*3];
      z0 += hv * w[0]; z1 += hv * w[1]; z2 += hv * w[2];
    }
    cf0 = z0 + __shfl_xor(z0, 16, 32) + i1_b2[0];
    cf1 = z1 + __shfl_xor(z1, 16, 32) + i1_b2[1];
    cf2 = z2 + __shfl_xor(z2, 16, 32) + i1_b2[2];
  }
  __syncthreads();

  wgemm_s<false, _Float16>(X, 128, 128, i2_w1, i2_b1, h, true, lane);
  __syncthreads();
  {
    float z0 = 0.f, z1 = 0.f, z2 = 0.f;
    for (int k = 0; k < 64; ++k) {
      float hv = (float)h[row*128 + half*64 + k];
      const float* w = &i2_w2[(half*64 + k)*3];
      z0 += hv * w[0]; z1 += hv * w[1]; z2 += hv * w[2];
    }
    ca0 = z0 + __shfl_xor(z0, 16, 32) + i2_b2[0];
    ca1 = z1 + __shfl_xor(z1, 16, 32) + i2_b2[1];
    ca2 = z2 + __shfl_xor(z2, 16, 32) + i2_b2[2];
  }
  __syncthreads();

  wgemm_s<false, _Float16>(X, 128, 128, fs_w1, fs_b1, h, true, lane);
  __syncthreads();
  {
    float z = 0.f;
    for (int k = 0; k < 64; ++k) z += (float)h[row*128 + half*64 + k] * fs_w2[half*64 + k];
    lam = z + __shfl_xor(z, 16, 32) + fs_b2[0];
  }

  if (half == 0 && row0 + row < E) {
    const int e = row0 + row;
    const int s = ei[e], r = ei[E + e];
    const float* bb = basis + (size_t)e * 9;
    F3 va{bb[0], bb[1], bb[2]}, vb{bb[3], bb[4], bb[5]}, vc{bb[6], bb[7], bb[8]};
    F3 fij = va * cf0 + vb * cf1 + vc * cf2;
    F3 aij = va * ca0 + vb * ca1 + vc * ca2;
    float wsv = w_sig[s], wrv = w_sig[r];
    F3 sp = ld3(pos, s), rp = ld3(pos, r);
    float winv = 1.f / (wsv + wrv + 1e-8f);
    F3 r0 = (sp * wsv + rp * wrv) * winv;
    F3 dd = rp - r0;
    F3 fl = fij * lam;
    F3 tau = aij - cross3(dd, fl);
    atomicAdd(&out_f[(size_t)r*3 + 0], fij.x);
    atomicAdd(&out_f[(size_t)r*3 + 1], fij.y);
    atomicAdd(&out_f[(size_t)r*3 + 2], fij.z);
    atomicAdd(&out_t[(size_t)r*3 + 0], tau.x);
    atomicAdd(&out_t[(size_t)r*3 + 1], tau.y);
    atomicAdd(&out_t[(size_t)r*3 + 2], tau.z);
  }
}

// ---------------- final combine ---------------------------------------------
__global__ void k_node_out(const float* __restrict__ m_inv, const float* __restrict__ i_inv,
                           const float* __restrict__ de_o,
                           const float* __restrict__ of, const float* __restrict__ ot,
                           float* __restrict__ dv, float* __restrict__ dw, int N) {
  int i = blockIdx.x * blockDim.x + threadIdx.x;
  if (i >= N * 3) return;
  int node = i / 3;
  dv[i] = m_inv[node] * of[i] + de_o[i];
  dw[i] = i_inv[node] * ot[i];
}

// ---------------------------------------------------------------------------
extern "C" void kernel_launch(void* const* d_in, const int* in_sizes, int n_in,
                              void* d_out, int out_size, void* d_ws, size_t ws_size,
                              hipStream_t stream) {
  const int N = in_sizes[1] / 3;   // pos is [N,3]
  const int E = in_sizes[9] / 3;   // edge_attr is [E,3]

  const int*   ei    = (const int*)d_in[0];
  const float* pos   = (const float*)d_in[1];
  const float* vel   = (const float*)d_in[2];
  const float* pvel  = (const float*)d_in[3];
  const float* omg   = (const float*)d_in[4];
  const float* pomg  = (const float*)d_in[5];
  const float* acl   = (const float*)d_in[6];
  const float* alp   = (const float*)d_in[7];
  const float* nfeat = (const float*)d_in[8];
  const float* eattr = (const float*)d_in[9];

  const float* ne_w1 = (const float*)d_in[10]; const float* ne_b1 = (const float*)d_in[11];
  const float* ne_w2 = (const float*)d_in[12]; const float* ne_b2 = (const float*)d_in[13];
  const float* ne_g  = (const float*)d_in[14]; const float* ne_b  = (const float*)d_in[15];
  const float* ef_w1 = (const float*)d_in[16]; const float* ef_b1 = (const float*)d_in[17];
  const float* ef_w2 = (const float*)d_in[18]; const float* ef_b2 = (const float*)d_in[19];
  const float* ef_g  = (const float*)d_in[20]; const float* ef_b  = (const float*)d_in[21];
  const float* ee_w1 = (const float*)d_in[22]; const float* ee_b1 = (const float*)d_in[23];
  const float* ee_w2 = (const float*)d_in[24]; const float* ee_b2 = (const float*)d_in[25];
  const float* ee_g  = (const float*)d_in[26]; const float* ee_b  = (const float*)d_in[27];
  const float* ie_w1 = (const float*)d_in[28]; const float* ie_b1 = (const float*)d_in[29];
  const float* ie_w2 = (const float*)d_in[30]; const float* ie_b2 = (const float*)d_in[31];
  const float* ie_g  = (const float*)d_in[32]; const float* ie_b  = (const float*)d_in[33];
  const float* i1_w1 = (const float*)d_in[34]; const float* i1_b1 = (const float*)d_in[35];
  const float* i1_w2 = (const float*)d_in[36]; const float* i1_b2 = (const float*)d_in[37];
  const float* i2_w1 = (const float*)d_in[38]; const float* i2_b1 = (const float*)d_in[39];
  const float* i2_w2 = (const float*)d_in[40]; const float* i2_b2 = (const float*)d_in[41];
  const float* fs_w1 = (const float*)d_in[42]; const float* fs_b1 = (const float*)d_in[43];
  const float* fs_w2 = (const float*)d_in[44]; const float* fs_b2 = (const float*)d_in[45];
  const float* nw_w1 = (const float*)d_in[46]; const float* nw_b1 = (const float*)d_in[47];
  const float* nw_w2 = (const float*)d_in[48]; const float* nw_b2 = (const float*)d_in[49];
  const float* mi_w1 = (const float*)d_in[50]; const float* mi_b1 = (const float*)d_in[51];
  const float* mi_w2 = (const float*)d_in[52]; const float* mi_b2 = (const float*)d_in[53];
  const float* ii_w1 = (const float*)d_in[54]; const float* ii_b1 = (const float*)d_in[55];
  const float* ii_w2 = (const float*)d_in[56]; const float* ii_b2 = (const float*)d_in[57];
  const float* de_w1 = (const float*)d_in[58]; const float* de_b1 = (const float*)d_in[59];
  const float* de_w2 = (const float*)d_in[60]; const float* de_b2 = (const float*)d_in[61];

  // output layout: node_dv [N,3] | node_dw [N,3] | il [E,128]
  float* dv_out = (float*)d_out;
  float* dw_out = dv_out + (size_t)3 * N;
  float* il_out = dv_out + (size_t)6 * N;

  // ---- workspace carve ----
  char* wp = (char*)d_ws;
  auto carve = [&](size_t bytes) -> void* {
    void* p = (void*)wp;
    wp += (bytes + 255) & ~(size_t)255;
    return p;
  };
  _Float16* ne_w1p = (_Float16*)carve(32  * 128 * 2);
  _Float16* ef_w1p = (_Float16*)carve(32  * 128 * 2);
  _Float16* ee_w1p = (_Float16*)carve(32  * 128 * 2);
  _Float16* ie_w1p = (_Float16*)carve(384 * 128 * 2);
  _Float16* ne_w2p = (_Float16*)carve(128 * 128 * 2);
  _Float16* ef_w2p = (_Float16*)carve(128 * 128 * 2);
  _Float16* ee_w2p = (_Float16*)carve(128 * 128 * 2);
  _Float16* ie_w2p = (_Float16*)carve(128 * 128 * 2);
  _Float16* i1_w1p = (_Float16*)carve(128 * 128 * 2);
  _Float16* i2_w1p = (_Float16*)carve(128 * 128 * 2);
  _Float16* fs_w1p = (_Float16*)carve(128 * 128 * 2);
  _Float16* nw_w1p = (_Float16*)carve(128 * 128 * 2);
  _Float16* mi_w1p = (_Float16*)carve(128 * 128 * 2);
  _Float16* ii_w1p = (_Float16*)carve(128 * 128 * 2);
  _Float16* de_w1p = (_Float16*)carve(128 * 128 * 2);
  _Float16* nl16a  = (_Float16*)carve((size_t)N * 128 * 2);
  _Float16* il16a  = (_Float16*)carve((size_t)E * 128 * 2);
  float*    basis  = (float*)   carve((size_t)E * 9 * 4);
  float*    w_sig  = (float*)   carve((size_t)N * 4);
  float*    m_inv  = (float*)   carve((size_t)N * 4);
  float*    i_inv  = (float*)   carve((size_t)N * 4);
  float*    de_o   = (float*)   carve((size_t)N * 3 * 4);
  float*    out_f  = (float*)   carve((size_t)N * 3 * 4);
  float*    out_t  = (float*)   carve((size_t)N * 3 * 4);

  // ---- weight packing (f32 row-major -> f16 B-fragment tiles) ----
  auto pack = [&](const float* src, _Float16* dst, int K, int Kpad) {
    int n = Kpad * 128;
    k_pack<<<(n + 255) / 256, 256, 0, stream>>>(src, dst, K, Kpad);
  };
  pack(ne_w1, ne_w1p, 6, 32);    pack(ef_w1, ef_w1p, 18, 32);
  pack(ee_w1, ee_w1p, 4, 32);    pack(ie_w1, ie_w1p, 384, 384);
  pack(ne_w2, ne_w2p, 128, 128); pack(ef_w2, ef_w2p, 128, 128);
  pack(ee_w2, ee_w2p, 128, 128); pack(ie_w2, ie_w2p, 128, 128);
  pack(i1_w1, i1_w1p, 128, 128); pack(i2_w1, i2_w1p, 128, 128);
  pack(fs_w1, fs_w1p, 128, 128); pack(nw_w1, nw_w1p, 128, 128);
  pack(mi_w1, mi_w1p, 128, 128); pack(ii_w1, ii_w1p, 128, 128);
  pack(de_w1, de_w1p, 128, 128);

  // ---- zero scatter accumulators (out_f, out_t contiguous) ----
  k_zero<<<(6 * N + 255) / 256, 256, 0, stream>>>(out_f, 6 * N);

  const int ntiles = (N + 15) / 16;
  const int etiles = (E + 15) / 16;

  k_node_latent<<<ntiles, 32, 0, stream>>>(nfeat, N, ne_w1p, ne_b1, ne_w2p, ne_b2,
                                           ne_g, ne_b, nl16a);

  k_node_head<<<ntiles, 32, 0, stream>>>(nl16a, N,
                                         nw_w1p, nw_b1, nw_w2, nw_b2,
                                         mi_w1p, mi_b1, mi_w2, mi_b2,
                                         ii_w1p, ii_b1, ii_w2, ii_b2,
                                         de_w1p, de_b1, de_w2, de_b2,
                                         w_sig, m_inv, i_inv, de_o);

  k_edge_enc<<<etiles, 32, 0, stream>>>(ei, E, N, pos, vel, pvel, omg, pomg, acl, alp, eattr,
                                        ef_w1p, ef_b1, ef_w2p, ef_b2, ef_g, ef_b,
                                        ee_w1p, ee_b1, ee_w2p, ee_b2, ee_g, ee_b,
                                        ie_w1p, ie_b1, ie_w2p, ie_b2, ie_g, ie_b,
                                        nl16a, basis, il_out, il16a);

  k_edge_dec<<<etiles, 32, 0, stream>>>(ei, E, il16a, pos, basis, w_sig,
                                        i1_w1p, i1_b1, i1_w2, i1_b2,
                                        i2_w1p, i2_b1, i2_w2, i2_b2,
                                        fs_w1p, fs_b1, fs_w2, fs_b2,
                                        out_f, out_t);

  k_node_out<<<(3 * N + 255) / 256, 256, 0, stream>>>(m_inv, i_inv, de_o,
                                                      out_f, out_t, dv_out, dw_out, N);
}
